// EdgeConv_69337952027034
// MI455X (gfx1250) — matrix operand order repaired and checked by reading, not compiled, not run on hardware
//
#include <hip/hip_runtime.h>

#define NBATCH 32
#define PP     1000
#define KNN    16
#define C0     64
#define NPTS   (NBATCH * PP)
#define BN_EPS 1e-3f
#define WAVES  4

typedef __attribute__((ext_vector_type(16))) __bf16 v16bf;
typedef __attribute__((ext_vector_type(8)))  float  v8f;

// Hardware f32->bf16 conversion (clang lowers casts to v_cvt_*bf16* on gfx1250,
// pairing adjacent converts into v_cvt_pk_bf16_f32 where possible).
static __device__ __forceinline__ __bf16 f2bf(float f) { return (__bf16)f; }
static __device__ __forceinline__ unsigned pack2(float a, float b) {
  __bf16 lo = (__bf16)a, hi = (__bf16)b;
  unsigned short ul, uh;
  __builtin_memcpy(&ul, &lo, 2);
  __builtin_memcpy(&uh, &hi, 2);
  return (unsigned)ul | ((unsigned)uh << 16);
}

// ---------------- KNN: one thread per (n,p), insertion-select K+1 smallest ----
__global__ void knn_kernel(const float* __restrict__ pts, int* __restrict__ idx) {
  int t = blockIdx.x * blockDim.x + threadIdx.x;
  if (t >= NPTS) return;
  int n = t / PP, p = t - n * PP;
  const float* base = pts + (size_t)n * PP * 3;
  float px = base[p * 3 + 0], py = base[p * 3 + 1], pz = base[p * 3 + 2];
  float bd[KNN + 1];
  int   bi[KNN + 1];
#pragma unroll
  for (int i = 0; i <= KNN; ++i) { bd[i] = 3.4e38f; bi[i] = 0; }
  for (int q = 0; q < PP; ++q) {
    float dx = base[q * 3 + 0] - px;
    float dy = base[q * 3 + 1] - py;
    float dz = base[q * 3 + 2] - pz;
    float d = dx * dx + dy * dy + dz * dz;
    if (d < bd[KNN]) {
      int j = KNN;
      while (j > 0 && bd[j - 1] > d) { bd[j] = bd[j - 1]; bi[j] = bi[j - 1]; --j; }
      bd[j] = d; bi[j] = q;
    }
  }
#pragma unroll
  for (int k = 0; k < KNN; ++k) idx[(size_t)t * KNN + k] = bi[k + 1]; // drop self
}

// Fragment-layout helpers (inverse of documented 16-bit A 16x32 VGPR layout):
// column c within a 32-wide kb block maps to lane-half 'hi' and half-index 'j':
//   hi = (c>>3)&1 ;  j = (c&7) + ((c>>4)<<3)
// Frag storage: halves [(frag*32 + lane)*16 + j], 32B contiguous per lane.

#define WMMA_BF16(A_, B_, C_) \
  __builtin_amdgcn_wmma_f32_16x16x32_bf16(false, (A_), false, (B_), (short)0, (C_), false, false)

static __device__ __forceinline__ v16bf ld_frag(const __bf16* base, int slot, int lane) {
  return *(const v16bf*)(base + ((size_t)(slot * 32 + lane) << 4));
}

// Stage weight matrix (Cin x 64, row-major f32) into fragment-ordered bf16 LDS.
// frag f = kb*4+nb ; lane L: n = nb*16+(L&15), halves j -> W[(kb*32+((L>>4)<<3)+(j&7)+((j>>3)<<4))*64+n]
static __device__ __forceinline__ void stage_w(const float* __restrict__ w,
                                               __bf16* dst, int total_halves, int tid) {
  unsigned* d32 = (unsigned*)dst;
  for (int i2 = tid; i2 < (total_halves >> 1); i2 += 128) {
    int idx  = i2 << 1;
    int f    = idx >> 9;            // 512 halves per frag
    int rem  = idx & 511;
    int lane = rem >> 4;
    int j    = rem & 15;            // even
    int kb = f >> 2, nb = f & 3;
    int n  = (nb << 4) + (lane & 15);
    int k0 = kb * 32 + ((lane >> 4) << 3) + (j & 7) + ((j >> 3) << 4);
    d32[i2] = pack2(w[k0 * 64 + n], w[(k0 + 1) * 64 + n]);
  }
}

// ---------------- EdgeConv main: one wave per point --------------------------
__global__ void __launch_bounds__(128)
edgeconv_kernel(const float* __restrict__ features, const int* __restrict__ idx,
                const float* __restrict__ w1, const float* __restrict__ w2,
                const float* __restrict__ w3,
                const float* __restrict__ g1, const float* __restrict__ b1,
                const float* __restrict__ m1, const float* __restrict__ vv1,
                const float* __restrict__ g2, const float* __restrict__ b2,
                const float* __restrict__ m2, const float* __restrict__ vv2,
                const float* __restrict__ g3, const float* __restrict__ b3,
                const float* __restrict__ m3, const float* __restrict__ vv3,
                const float* __restrict__ scw, const float* __restrict__ scg,
                const float* __restrict__ scb, const float* __restrict__ scm,
                const float* __restrict__ scv,
                float* __restrict__ out) {
  // All tiles live in LDS in WMMA fragment order (32B contiguous per lane per frag)
  __shared__ __attribute__((aligned(32))) __bf16 sW1[16 * 32 * 16]; // 16 frags
  __shared__ __attribute__((aligned(32))) __bf16 sW2[8 * 32 * 16];  //  8 frags
  __shared__ __attribute__((aligned(32))) __bf16 sW3[8 * 32 * 16];  //  8 frags
  __shared__ __attribute__((aligned(32))) __bf16 sA[WAVES][4 * 32 * 16]; // 4 kb frags
  __shared__ __attribute__((aligned(16))) float  sFc[WAVES][64];

  int tid = threadIdx.x;
  stage_w(w1, sW1, 128 * 64, tid);
  stage_w(w2, sW2,  64 * 64, tid);
  stage_w(w3, sW3,  64 * 64, tid);

  int w    = tid >> 5;
  int lane = tid & 31;
  int pt   = blockIdx.x * WAVES + w;        // grid exactly covers NPTS
  int n    = pt / PP;

  const float* fcen = features + (size_t)pt * C0;
  float c0 = fcen[lane], c1 = fcen[lane + 32];
  sFc[w][lane] = c0; sFc[w][lane + 32] = c1;

  __bf16* A = sA[w];
  // swizzle coords for this lane's staged columns (c_local == lane for all 4 kb)
  int hi = (lane >> 3) & 1;
  int jj = (lane & 7) + ((lane >> 4) << 3);

  const int* my_idx = idx + (size_t)pt * KNN;
#pragma unroll 4
  for (int k = 0; k < KNN; ++k) {
    int q = my_idx[k];
    const float* fn = features + ((size_t)n * PP + q) * C0;
    float n0 = fn[lane], n1 = fn[lane + 32];
    int ldst = (k + (hi << 4)) << 4;   // (m + hi*16)*16
    A[(0 * 512) + ldst + jj] = f2bf(c0);
    A[(1 * 512) + ldst + jj] = f2bf(c1);
    A[(2 * 512) + ldst + jj] = f2bf(n0 - c0);
    A[(3 * 512) + ldst + jj] = f2bf(n1 - c1);
  }
  __syncthreads();

  int ln15 = lane & 15;
  int mrow = (lane >> 4) << 3;            // row base held by this lane in D frags
  int hi_o = (ln15 >> 3) & 1;             // writeback swizzle (depends on ln15 only)

  // ---- Layer 1: (16x128) @ (128x64) ----
  v8f acc[4] = {};
#pragma unroll
  for (int kb = 0; kb < 4; ++kb) {
    v16bf a = ld_frag(A, kb, lane);
#pragma unroll
    for (int nb = 0; nb < 4; ++nb)
      acc[nb] = WMMA_BF16(a, ld_frag(sW1, kb * 4 + nb, lane), acc[nb]);
  }
  // BN + ReLU, write back as next layer's A frags (16x64 -> kb 0..1)
#pragma unroll
  for (int nb = 0; nb < 4; ++nb) {
    int ch = (nb << 4) + ln15;
    float s = g1[ch] * rsqrtf(vv1[ch] + BN_EPS);
    float t = b1[ch] - m1[ch] * s;
    int jo  = (ln15 & 7) + ((nb & 1) << 3);
    int kbD = nb >> 1;
#pragma unroll
    for (int r = 0; r < 8; ++r) {
      float y = acc[nb][r] * s + t;
      y = y > 0.f ? y : 0.f;
      A[kbD * 512 + ((r + mrow + (hi_o << 4)) << 4) + jo] = f2bf(y);
    }
  }

  // ---- Layer 2: (16x64) @ (64x64) ----
  v8f acc2[4] = {};
#pragma unroll
  for (int kb = 0; kb < 2; ++kb) {
    v16bf a = ld_frag(A, kb, lane);
#pragma unroll
    for (int nb = 0; nb < 4; ++nb)
      acc2[nb] = WMMA_BF16(a, ld_frag(sW2, kb * 4 + nb, lane), acc2[nb]);
  }
#pragma unroll
  for (int nb = 0; nb < 4; ++nb) {
    int ch = (nb << 4) + ln15;
    float s = g2[ch] * rsqrtf(vv2[ch] + BN_EPS);
    float t = b2[ch] - m2[ch] * s;
    int jo  = (ln15 & 7) + ((nb & 1) << 3);
    int kbD = nb >> 1;
#pragma unroll
    for (int r = 0; r < 8; ++r) {
      float y = acc2[nb][r] * s + t;
      y = y > 0.f ? y : 0.f;
      A[kbD * 512 + ((r + mrow + (hi_o << 4)) << 4) + jo] = f2bf(y);
    }
  }

  // ---- Layer 3: (16x64) @ (64x64), fused BN+ReLU+mean over K rows ----
  v8f acc3[4] = {};
#pragma unroll
  for (int kb = 0; kb < 2; ++kb) {
    v16bf a = ld_frag(A, kb, lane);
#pragma unroll
    for (int nb = 0; nb < 4; ++nb)
      acc3[nb] = WMMA_BF16(a, ld_frag(sW3, kb * 4 + nb, lane), acc3[nb]);
  }
  float cs[4];
#pragma unroll
  for (int nb = 0; nb < 4; ++nb) {
    int ch = (nb << 4) + ln15;
    float s = g3[ch] * rsqrtf(vv3[ch] + BN_EPS);
    float t = b3[ch] - m3[ch] * s;
    float ps = 0.f;
#pragma unroll
    for (int r = 0; r < 8; ++r) {
      float y = acc3[nb][r] * s + t;
      y = y > 0.f ? y : 0.f;
      ps += y;
    }
    // lanes L and L^16 hold the same column (rows 0-7 vs 8-15)
    float tot = ps + __shfl_xor(ps, 16);
    cs[nb] = tot * (1.0f / (float)KNN);
  }

  // ---- Shortcut GEMV (1x64)@(64x64) + BN + add + ReLU ----
  float a0 = 0.f, a1 = 0.f;
#pragma unroll 4
  for (int c = 0; c < 64; c += 4) {
    float4 fc = *(const float4*)&sFc[w][c];
    a0 = fmaf(fc.x, scw[(c + 0) * 64 + lane], a0);
    a1 = fmaf(fc.x, scw[(c + 0) * 64 + lane + 32], a1);
    a0 = fmaf(fc.y, scw[(c + 1) * 64 + lane], a0);
    a1 = fmaf(fc.y, scw[(c + 1) * 64 + lane + 32], a1);
    a0 = fmaf(fc.z, scw[(c + 2) * 64 + lane], a0);
    a1 = fmaf(fc.z, scw[(c + 2) * 64 + lane + 32], a1);
    a0 = fmaf(fc.w, scw[(c + 3) * 64 + lane], a0);
    a1 = fmaf(fc.w, scw[(c + 3) * 64 + lane + 32], a1);
  }
  int d0 = lane, d1 = lane + 32;
  float s0 = scg[d0] * rsqrtf(scv[d0] + BN_EPS);
  float s1 = scg[d1] * rsqrtf(scv[d1] + BN_EPS);
  float r0 = a0 * s0 + (scb[d0] - scm[d0] * s0);
  float r1 = a1 * s1 + (scb[d1] - scm[d1] * s1);
  // channel 'lane' -> cs[0]/cs[1]; channel 'lane+32' -> cs[2]/cs[3]
  float f0 = (lane & 16) ? cs[1] : cs[0];
  float f1 = (lane & 16) ? cs[3] : cs[2];
  float o0 = r0 + f0; o0 = o0 > 0.f ? o0 : 0.f;
  float o1 = r1 + f1; o1 = o1 > 0.f ? o1 : 0.f;
  out[(size_t)pt * 64 + lane]      = o0;
  out[(size_t)pt * 64 + lane + 32] = o1;
}

extern "C" void kernel_launch(void* const* d_in, const int* in_sizes, int n_in,
                              void* d_out, int out_size, void* d_ws, size_t ws_size,
                              hipStream_t stream) {
  (void)in_sizes; (void)n_in; (void)out_size; (void)ws_size;
  const float* points   = (const float*)d_in[0];
  const float* features = (const float*)d_in[1];
  const float* w1 = (const float*)d_in[2];
  const float* w2 = (const float*)d_in[3];
  const float* w3 = (const float*)d_in[4];
  const float* g1 = (const float*)d_in[5];
  const float* g2 = (const float*)d_in[6];
  const float* g3 = (const float*)d_in[7];
  const float* b1 = (const float*)d_in[8];
  const float* b2 = (const float*)d_in[9];
  const float* b3 = (const float*)d_in[10];
  const float* m1 = (const float*)d_in[11];
  const float* m2 = (const float*)d_in[12];
  const float* m3 = (const float*)d_in[13];
  const float* v1 = (const float*)d_in[14];
  const float* v2 = (const float*)d_in[15];
  const float* v3 = (const float*)d_in[16];
  const float* scw = (const float*)d_in[17];
  const float* scg = (const float*)d_in[18];
  const float* scb = (const float*)d_in[19];
  const float* scm = (const float*)d_in[20];
  const float* scv = (const float*)d_in[21];
  float* out = (float*)d_out;
  int*   idx = (int*)d_ws;   // N*P*K ints = 2 MB scratch

  knn_kernel<<<(NPTS + 255) / 256, 256, 0, stream>>>(points, idx);
  edgeconv_kernel<<<NPTS / WAVES, 128, 0, stream>>>(
      features, idx, w1, w2, w3,
      g1, b1, m1, v1, g2, b2, m2, v2, g3, b3, m3, v3,
      scw, scg, scb, scm, scv, out);
}
